// MemoryEfficientS6Compressor_35682588295478
// MI455X (gfx1250) — matrix operand order repaired and checked by reading, not compiled, not run on hardware
//
#include <hip/hip_runtime.h>
#include <math.h>

typedef __attribute__((ext_vector_type(16))) _Float16 v16h;
typedef __attribute__((ext_vector_type(8)))  float    v8f;

// ---------------- problem constants (last-chunk-only schedule) ----------------
// Only chunk 3 (global t 96..127) feeds the output (ys[-8:]).
//   xi  needed at chunk pos 14..31  -> 18 steps  (rows 1152)
//   xc  needed at chunk pos 17..31  -> 15 steps  (rows 960)
//   z,dt,ys at chunk pos 24..31     -> 8 steps   (rows 512)
#define DINNER 2457
#define DPAD   2464          // DINNER padded to mult of 32
#define NPADW  2496          // W_in/W_gate row pad (mult of 64)
#define DMODEL 2048
#define BATCH  64
#define KCONV  1408          // 4 taps * 352 (351 in-ch per group + 1 zero pad)
#define OPADG  384           // 351 out-ch per group padded to mult of 64

// ---------------- scalar helper kernels ----------------

__global__ void cvt_pad(const float* __restrict__ src, _Float16* __restrict__ dst,
                        int srcR, int srcC, int dstR, int dstC) {
  size_t i = (size_t)blockIdx.x * blockDim.x + threadIdx.x;
  size_t tot = (size_t)dstR * dstC;
  if (i >= tot) return;
  int r = (int)(i / dstC), c = (int)(i % dstC);
  float v = (r < srcR && c < srcC) ? src[(size_t)r * srcC + c] : 0.f;
  dst[i] = (_Float16)v;
}

// W_conv [2457][351][4] f32  ->  [7][384][1408] f16, K packed as (k,i): c = k*352 + i
__global__ void pack_wconv(const float* __restrict__ W, _Float16* __restrict__ dst) {
  size_t i = (size_t)blockIdx.x * blockDim.x + threadIdx.x;
  const size_t tot = (size_t)7 * OPADG * KCONV;
  if (i >= tot) return;
  int c = (int)(i % KCONV);
  int o = (int)((i / KCONV) % OPADG);
  int g = (int)(i / ((size_t)KCONV * OPADG));
  int k = c / 352, ii = c % 352;
  float v = 0.f;
  if (o < 351 && ii < 351)
    v = W[((size_t)(g * 351 + o) * 351 + ii) * 4 + k];
  dst[i] = (_Float16)v;
}

// im2col for the grouped causal conv: dst[7][960][1408]
// row r=(t_rel,b), t_rel 0..14 maps to xc position t=17+t_rel; col c=(k,i).
// source xi16 stored [18][64][DPAD] for xi positions 14..31; index = t_rel + k.
__global__ void im2col_k(const _Float16* __restrict__ xi16, _Float16* __restrict__ dst) {
  size_t i = (size_t)blockIdx.x * blockDim.x + threadIdx.x;
  const size_t tot = (size_t)7 * 960 * KCONV;
  if (i >= tot) return;
  int c = (int)(i % KCONV);
  int r = (int)((i / KCONV) % 960);
  int g = (int)(i / ((size_t)KCONV * 960));
  int k = c / 352, ii = c % 352;
  _Float16 v = (_Float16)0.f;
  if (ii < 351) {
    int t_rel = r >> 6, b = r & 63;
    v = xi16[(size_t)((t_rel + k) * BATCH + b) * DPAD + g * 351 + ii];
  }
  dst[i] = v;
}

// xp[r][j] = b_xproj[j] + dot(xc_row, W_xproj[j][0..2456]);  r=(tt,b) tt 0..7 -> xc row 7+tt
__global__ void xproj_k(const float* __restrict__ xc, const float* __restrict__ Wx,
                        const float* __restrict__ bx, float* __restrict__ xp) {
  int idx = blockIdx.x * blockDim.x + threadIdx.x;
  if (idx >= 512 * 32) return;
  int r = idx >> 5, j = idx & 31;
  int tt = r >> 6, b = r & 63;
  const float* xrow = xc + (size_t)((7 + tt) * BATCH + b) * DPAD;
  const float* wrow = Wx + (size_t)j * DINNER;
  float s = bx[j];
  for (int d = 0; d < DINNER; ++d) s += xrow[d] * wrow[d];
  xp[idx] = s;
}

// dt[r][n] = softplus(xp_row . W_dt[n]) + 1e-4, stored at stride DPAD
__global__ void dt_k(const float* __restrict__ xp, const float* __restrict__ Wdt,
                     float* __restrict__ dt) {
  size_t i = (size_t)blockIdx.x * blockDim.x + threadIdx.x;
  const size_t tot = (size_t)512 * DINNER;
  if (i >= tot) return;
  int r = (int)(i / DINNER), n = (int)(i % DINNER);
  const float* xr = xp + (size_t)r * 32;
  const float* wr = Wdt + (size_t)n * 32;
  float s = 0.f;
  #pragma unroll
  for (int j = 0; j < 32; ++j) s += xr[j] * wr[j];
  float sp = (s > 20.f) ? s : log1pf(expf(s));
  dt[(size_t)r * DPAD + n] = sp + 1e-4f;
}

// window softmax mix + skip + gate, averaged over the final 8 steps -> compressed (f16, padded)
__global__ void mix_k(const float* __restrict__ xc, const float* __restrict__ dt,
                      const float* __restrict__ sigz, const float* __restrict__ Dparam,
                      _Float16* __restrict__ comp16) {
  size_t i = (size_t)blockIdx.x * blockDim.x + threadIdx.x;
  if (i >= (size_t)BATCH * DPAD) return;
  int b = (int)(i / DPAD), d = (int)(i % DPAD);
  if (d >= DINNER) { comp16[i] = (_Float16)0.f; return; }
  float xv[15];
  #pragma unroll
  for (int j = 0; j < 15; ++j) xv[j] = xc[(size_t)(j * BATCH + b) * DPAD + d];
  float Dp = Dparam[d];
  float accum = 0.f;
  #pragma unroll
  for (int tt = 0; tt < 8; ++tt) {
    float dtv = dt[(size_t)(tt * BATCH + b) * DPAD + d];
    float se = 0.f, sx = 0.f;
    #pragma unroll
    for (int k = 0; k < 8; ++k) {            // softmax over window, stabilized (dtv>0)
      float e = expf(dtv * (float)(k - 7));
      se += e; sx += e * xv[tt + k];
    }
    float local = sx / se;
    float g = sigz[(size_t)(tt * BATCH + b) * DPAD + d];
    accum += (local + Dp * xv[tt + 7]) * g;
  }
  comp16[i] = (_Float16)(accum * 0.125f);
}

// in-place layernorm over rows of [64][2048]
__global__ void ln_k(float* __restrict__ out, const float* __restrict__ lnw,
                     const float* __restrict__ lnb) {
  __shared__ float sh[512];
  int b = blockIdx.x, tid = threadIdx.x;           // 256 threads
  float v[8], s = 0.f, s2 = 0.f;
  #pragma unroll
  for (int j = 0; j < 8; ++j) {
    v[j] = out[(size_t)b * DMODEL + tid + j * 256];
    s += v[j]; s2 += v[j] * v[j];
  }
  sh[tid] = s; sh[256 + tid] = s2;
  __syncthreads();
  for (int off = 128; off > 0; off >>= 1) {
    if (tid < off) { sh[tid] += sh[tid + off]; sh[256 + tid] += sh[256 + tid + off]; }
    __syncthreads();
  }
  float mu  = sh[0] * (1.f / DMODEL);
  float var = sh[256] * (1.f / DMODEL) - mu * mu;
  float inv = rsqrtf(var + 1e-5f);
  #pragma unroll
  for (int j = 0; j < 8; ++j) {
    int c = tid + j * 256;
    out[(size_t)b * DMODEL + c] = (v[j] - mu) * inv * lnw[c] + lnb[c];
  }
}

// ---------------- WMMA GEMM:  C[M][N] = act( A[M][K] * B[N][K]^T + bias[N] ) --------
// 128 threads = 4 waves; 64x64 block tile; 32x32 per wave (2x2 wmma frags); K step 32.
// act: 0=none 1=silu 2=sigmoid.  outF16: write _Float16 else float.  All M mult of 64,
// B row-padded to grid, K mult of 32, so the hot loop has EXEC all-ones (WMMA-legal).
__global__ __launch_bounds__(128) void gemm_wmma(
    const _Float16* __restrict__ A, int lda,
    const _Float16* __restrict__ B, int ldb,
    const float* __restrict__ bias,
    void* __restrict__ Cout, int ldc,
    int M, int N, int K, int act, int outF16)
{
  __shared__ __align__(16) _Float16 sA[64 * 40];   // row stride 40 halves (80B, 16B-aligned)
  __shared__ __align__(16) _Float16 sB[64 * 40];

  const int tid  = threadIdx.x;
  const int wave = tid >> 5, lane = tid & 31;
  const int wm = wave >> 1, wn = wave & 1;
  const int hh = lane >> 4, rr = lane & 15;
  const int m0 = blockIdx.y * 64, n0 = blockIdx.x * 64;
  const int lrow = tid >> 1, lseg = tid & 1;       // 128 thr: 64 rows x 2 segs of 16 halves

  v8f acc[2][2] = {};

  const _Float16* gA = A + (size_t)(m0 + lrow) * lda + lseg * 16;
  const _Float16* gB = B + (size_t)(n0 + lrow) * ldb + lseg * 16;
  _Float16* lA = &sA[lrow * 40 + lseg * 16];
  _Float16* lB = &sB[lrow * 40 + lseg * 16];

  for (int k0 = 0; k0 < K; k0 += 32) {
    uint4 va0 = *reinterpret_cast<const uint4*>(gA + k0);
    uint4 va1 = *reinterpret_cast<const uint4*>(gA + k0 + 8);
    uint4 vb0 = *reinterpret_cast<const uint4*>(gB + k0);
    uint4 vb1 = *reinterpret_cast<const uint4*>(gB + k0 + 8);
    if (k0 + 32 < K) {
      __builtin_prefetch(gA + k0 + 32, 0, 1);
      __builtin_prefetch(gB + k0 + 32, 0, 1);
    }
    __syncthreads();
    reinterpret_cast<uint4*>(lA)[0] = va0;
    reinterpret_cast<uint4*>(lA)[1] = va1;
    reinterpret_cast<uint4*>(lB)[0] = vb0;
    reinterpret_cast<uint4*>(lB)[1] = vb1;
    __syncthreads();

    v16h af[2], bf[2];
    #pragma unroll
    for (int mi = 0; mi < 2; ++mi) {
      // A 16x32 layout: lane(h,r): K = {8h..8h+7, 16+8h..23+8h} of row r
      const _Float16* pa = &sA[(wm * 32 + mi * 16 + rr) * 40];
      reinterpret_cast<uint4*>(&af[mi])[0] = *reinterpret_cast<const uint4*>(pa + 8 * hh);
      reinterpret_cast<uint4*>(&af[mi])[1] = *reinterpret_cast<const uint4*>(pa + 16 + 8 * hh);
    }
    #pragma unroll
    for (int ni = 0; ni < 2; ++ni) {
      // B 32x16 layout: lane(h,n): K = {8h..8h+7, 16+8h..23+8h} of column n
      const _Float16* pb = &sB[(wn * 32 + ni * 16 + rr) * 40];
      reinterpret_cast<uint4*>(&bf[ni])[0] = *reinterpret_cast<const uint4*>(pb + 8 * hh);
      reinterpret_cast<uint4*>(&bf[ni])[1] = *reinterpret_cast<const uint4*>(pb + 16 + 8 * hh);
    }
    #pragma unroll
    for (int mi = 0; mi < 2; ++mi)
      #pragma unroll
      for (int ni = 0; ni < 2; ++ni)
        acc[mi][ni] = __builtin_amdgcn_wmma_f32_16x16x32_f16(
            false, af[mi], false, bf[ni], (short)0, acc[mi][ni], false, false);
  }

  // epilogue: C lane(h,n) VGPR v -> M = v + 8h
  #pragma unroll
  for (int mi = 0; mi < 2; ++mi) {
    #pragma unroll
    for (int ni = 0; ni < 2; ++ni) {
      int n = n0 + wn * 32 + ni * 16 + rr;
      if (n >= N) continue;
      float bv = bias ? bias[n] : 0.f;
      #pragma unroll
      for (int v = 0; v < 8; ++v) {
        int m = m0 + wm * 32 + mi * 16 + v + 8 * hh;
        if (m >= M) continue;
        float val = acc[mi][ni][v] + bv;
        if (act == 1)      val = val / (1.f + expf(-val));   // silu
        else if (act == 2) val = 1.f / (1.f + expf(-val));   // sigmoid
        if (outF16) ((_Float16*)Cout)[(size_t)m * ldc + n] = (_Float16)val;
        else        ((float*)Cout)[(size_t)m * ldc + n] = val;
      }
    }
  }
}

// ---------------- launch ----------------

static inline unsigned b1d(size_t n, int t) { return (unsigned)((n + t - 1) / t); }

extern "C" void kernel_launch(void* const* d_in, const int* in_sizes, int n_in,
                              void* d_out, int out_size, void* d_ws, size_t ws_size,
                              hipStream_t stream) {
  (void)in_sizes; (void)n_in; (void)out_size; (void)ws_size;

  const float* x       = (const float*)d_in[0];
  const float* W_in    = (const float*)d_in[1];
  const float* b_in    = (const float*)d_in[2];
  const float* W_gate  = (const float*)d_in[3];
  const float* b_gate  = (const float*)d_in[4];
  const float* W_conv  = (const float*)d_in[5];
  const float* b_conv  = (const float*)d_in[6];
  const float* W_xproj = (const float*)d_in[7];
  const float* b_xproj = (const float*)d_in[8];
  const float* W_dt    = (const float*)d_in[9];
  const float* Dparam  = (const float*)d_in[10];
  const float* W_out   = (const float*)d_in[11];
  const float* b_out   = (const float*)d_in[12];
  const float* ln_w    = (const float*)d_in[13];
  const float* ln_b    = (const float*)d_in[14];

  char* ws = (char*)d_ws;
  size_t off = 0;
  auto alloc = [&](size_t bytes) -> char* {
    char* p = ws + off; off += (bytes + 255) & ~(size_t)255; return p;
  };

  _Float16* Win16   = (_Float16*)alloc((size_t)NPADW * DMODEL * 2);
  _Float16* Wgate16 = (_Float16*)alloc((size_t)NPADW * DMODEL * 2);
  _Float16* Wout16  = (_Float16*)alloc((size_t)DMODEL * DPAD * 2);
  _Float16* X16     = (_Float16*)alloc((size_t)1152 * DMODEL * 2);
  _Float16* xi16    = (_Float16*)alloc((size_t)1152 * DPAD * 2);
  float*    sigz    = (float*)   alloc((size_t)512 * DPAD * 4);
  _Float16* Wc16    = (_Float16*)alloc((size_t)7 * OPADG * KCONV * 2);
  _Float16* col16   = (_Float16*)alloc((size_t)7 * 960 * KCONV * 2);
  float*    xcbuf   = (float*)   alloc((size_t)960 * DPAD * 4);
  float*    xpbuf   = (float*)   alloc((size_t)512 * 32 * 4);
  float*    dtbuf   = (float*)   alloc((size_t)512 * DPAD * 4);
  _Float16* comp16  = (_Float16*)alloc((size_t)BATCH * DPAD * 2);

  const int T = 256;
  // weight / input conversions (fp32 -> f16, zero-padded)
  cvt_pad<<<b1d((size_t)NPADW * DMODEL, T), T, 0, stream>>>(W_in,   Win16,   DINNER, DMODEL, NPADW, DMODEL);
  cvt_pad<<<b1d((size_t)NPADW * DMODEL, T), T, 0, stream>>>(W_gate, Wgate16, DINNER, DMODEL, NPADW, DMODEL);
  cvt_pad<<<b1d((size_t)DMODEL * DPAD,  T), T, 0, stream>>>(W_out,  Wout16,  DMODEL, DINNER, DMODEL, DPAD);
  cvt_pad<<<b1d((size_t)1152 * DMODEL,  T), T, 0, stream>>>(x + (size_t)110 * BATCH * DMODEL,
                                                            X16, 1152, DMODEL, 1152, DMODEL);
  pack_wconv<<<b1d((size_t)7 * OPADG * KCONV, T), T, 0, stream>>>(W_conv, Wc16);

  dim3 blk(128);
  // xi = X @ W_in^T + b_in   (positions 14..31) -> f16
  gemm_wmma<<<dim3(39, 18), blk, 0, stream>>>(X16, DMODEL, Win16, DMODEL, b_in,
                                              xi16, DPAD, 1152, DINNER, DMODEL, 0, 1);
  // sigz = sigmoid(X @ W_gate^T + b_gate)  (positions 24..31) -> f32
  gemm_wmma<<<dim3(39, 8), blk, 0, stream>>>(X16 + (size_t)640 * DMODEL, DMODEL, Wgate16, DMODEL,
                                             b_gate, sigz, DPAD, 512, DINNER, DMODEL, 2, 0);
  // grouped causal conv as 7 WMMA GEMMs through im2col; xc = silu(conv + b_conv)
  im2col_k<<<b1d((size_t)7 * 960 * KCONV, T), T, 0, stream>>>(xi16, col16);
  for (int g = 0; g < 7; ++g)
    gemm_wmma<<<dim3(6, 15), blk, 0, stream>>>(col16 + (size_t)g * 960 * KCONV, KCONV,
                                               Wc16 + (size_t)g * OPADG * KCONV, KCONV,
                                               b_conv + g * 351, xcbuf + g * 351, DPAD,
                                               960, 351, KCONV, 1, 0);
  // dt path (tiny): xproj (only first 32 cols are consumed) then softplus projection
  xproj_k<<<b1d(512 * 32, T), T, 0, stream>>>(xcbuf, W_xproj, b_xproj, xpbuf);
  dt_k<<<b1d((size_t)512 * DINNER, T), T, 0, stream>>>(xpbuf, W_dt, dtbuf);
  // window softmax mix + D-skip + gate, mean over last 8 steps -> compressed (f16)
  mix_k<<<b1d((size_t)BATCH * DPAD, T), T, 0, stream>>>(xcbuf, dtbuf, sigz, Dparam, comp16);
  // out = compressed @ W_out^T + b_out  -> d_out, then layernorm in place
  gemm_wmma<<<dim3(32, 1), blk, 0, stream>>>(comp16, DPAD, Wout16, DPAD, b_out,
                                             d_out, DMODEL, BATCH, DMODEL, DPAD, 0, 0);
  ln_k<<<BATCH, 256, 0, stream>>>((float*)d_out, ln_w, ln_b);
}
